// DACA_29343216566277
// MI455X (gfx1250) — compile-verified, matrix operand order checked by reading, not run on hardware
//
#include <hip/hip_runtime.h>
#include <hip/hip_bf16.h>
#include <cstdint>

// Problem constants (match reference)
#define B_  32
#define S_  512
#define D_  768
#define H_  8
#define DK_ 96
#define MEM_ 768

typedef __attribute__((ext_vector_type(16))) __bf16 v16bf;
typedef __attribute__((ext_vector_type(8)))  float  v8f;

__device__ __forceinline__ unsigned short f2bf(float f) {
  unsigned int u = __float_as_uint(f);
  unsigned int r = u + 0x7FFFu + ((u >> 16) & 1u);   // round-to-nearest-even
  return (unsigned short)(r >> 16);
}
__device__ __forceinline__ float bf2f(unsigned short h) {
  return __uint_as_float(((unsigned int)h) << 16);
}

__device__ __forceinline__ float block_reduce_sum(float v, float* red) {
  const int tid = threadIdx.x;
  red[tid] = v; __syncthreads();
  for (int s = 128; s > 0; s >>= 1) { if (tid < s) red[tid] += red[tid + s]; __syncthreads(); }
  float r = red[0]; __syncthreads();
  return r;
}

// ---------------------------------------------------------------------------
// Generic batched GEMM:  C[z] = act( (A[z] * B[z]^layout) * alpha + bias ) / denom
//   A: bf16 row-major [M,K]   (lda, strideAz in elements)
//   B: bf16; bNMajor=1 -> B[n*ldb+k] (i.e. W[N,K], compute A*W^T)
//            bNMajor=0 -> B[k*ldb+n] (plain K-major [K,N])
//   C: f32 or bf16 (cIsBf16), ldc/strideCz in elements
//
//   Block tile 128(M) x 128(N), 8 waves (wave32); each wave owns 32x64:
//   2 A-frags x 4 B-frags -> 8 v_wmma_f32_16x16x32_bf16 per 32-K step.
//   Double-buffered LDS (40 KB) -> one barrier per K-step; next tile's
//   global_load_b128/ds_store overlap the current tile's WMMA chain.
// ---------------------------------------------------------------------------
union FragBF { v16bf v; unsigned int u[8]; };

__global__ __launch_bounds__(256) void gemm_bf16_wmma(
    const unsigned short* __restrict__ A, long long lda, long long strideAz,
    const unsigned short* __restrict__ Bm, long long ldb, long long strideBz, int bNMajor,
    void* __restrict__ Cv, long long ldc, long long strideCz, int cIsBf16,
    const float* __restrict__ bias, const float* __restrict__ denom,
    int K, float alpha, int relu)
{
  __shared__ __align__(16) unsigned short As[2][128][40];   // 2 x 128x32 (+8 pad)
  __shared__ __align__(16) unsigned short Bs[2][128][40];   // transposed: [n][k]

  const int tid  = threadIdx.x;
  const int w    = tid >> 5;          // wave id 0..7
  const int wm   = w >> 1;            // 0..3 -> 32-row band
  const int wn   = w & 1;             // 0..1 -> 64-col band
  const int lane = tid & 31;
  const int m0   = blockIdx.y * 128;
  const int n0   = blockIdx.x * 128;

  const unsigned short* Ab = A  + (size_t)blockIdx.z * (size_t)strideAz;
  const unsigned short* Bb = Bm + (size_t)blockIdx.z * (size_t)strideBz;

  v8f acc[2][4];
  for (int a2 = 0; a2 < 2; ++a2)
    for (int i = 0; i < 4; ++i)
      for (int j = 0; j < 8; ++j) acc[a2][i][j] = 0.0f;

  const int ar = tid >> 1;            // stage: row 0..127
  const int ah = (tid & 1) * 16;      // stage: col half (0/16)

  auto stage = [&](int buf, int kt) {
    {
      const unsigned short* src = Ab + (size_t)(m0 + ar) * (size_t)lda + kt + ah;
      *(uint4*)&As[buf][ar][ah]     = *(const uint4*)(src);
      *(uint4*)&As[buf][ar][ah + 8] = *(const uint4*)(src + 8);
      __builtin_prefetch(src + 64, 0, 1);   // global_prefetch for K-tile after next
    }
    if (bNMajor) {
      const unsigned short* src = Bb + (size_t)(n0 + ar) * (size_t)ldb + kt + ah;
      *(uint4*)&Bs[buf][ar][ah]     = *(const uint4*)(src);
      *(uint4*)&Bs[buf][ar][ah + 8] = *(const uint4*)(src + 8);
    } else {
      const int k  = tid >> 3;            // 0..31
      const int oc = (tid & 7) * 16;      // 16 columns per thread
      const unsigned short* src = Bb + (size_t)(kt + k) * (size_t)ldb + n0 + oc;
      uint4 q0 = *(const uint4*)(src);
      uint4 q1 = *(const uint4*)(src + 8);
      __align__(16) unsigned short tmp[16];
      *(uint4*)tmp       = q0;
      *(uint4*)(tmp + 8) = q1;
      for (int j = 0; j < 16; ++j) Bs[buf][oc + j][k] = tmp[j];
    }
  };

  stage(0, 0);
  __syncthreads();

  int buf = 0;
  for (int kt = 0; kt < K; kt += 32) {
    if (kt + 32 < K) stage(buf ^ 1, kt + 32);   // overlap next stage with math

    // ---- fragments per ISA 7.12.2: lanes 0-15 K {lo..lo+7,16+lo..}, lanes16-31 lo=8 ----
    const int lo = (lane >> 4) * 8;
    const int r0 = wm * 32 + (lane & 15);
    FragBF fa[2], fb[4];
    for (int ms = 0; ms < 2; ++ms) {
      uint4 a0 = *(const uint4*)&As[buf][r0 + ms * 16][lo];
      uint4 a1 = *(const uint4*)&As[buf][r0 + ms * 16][16 + lo];
      fa[ms].u[0] = a0.x; fa[ms].u[1] = a0.y; fa[ms].u[2] = a0.z; fa[ms].u[3] = a0.w;
      fa[ms].u[4] = a1.x; fa[ms].u[5] = a1.y; fa[ms].u[6] = a1.z; fa[ms].u[7] = a1.w;
    }
    for (int ns = 0; ns < 4; ++ns) {
      const int n = wn * 64 + ns * 16 + (lane & 15);
      uint4 b0 = *(const uint4*)&Bs[buf][n][lo];
      uint4 b1 = *(const uint4*)&Bs[buf][n][16 + lo];
      fb[ns].u[0] = b0.x; fb[ns].u[1] = b0.y; fb[ns].u[2] = b0.z; fb[ns].u[3] = b0.w;
      fb[ns].u[4] = b1.x; fb[ns].u[5] = b1.y; fb[ns].u[6] = b1.z; fb[ns].u[7] = b1.w;
    }
#if defined(__HIP_DEVICE_COMPILE__)
    for (int ms = 0; ms < 2; ++ms)
      for (int ns = 0; ns < 4; ++ns)
        acc[ms][ns] = __builtin_amdgcn_wmma_f32_16x16x32_bf16(
            false, fa[ms].v, false, fb[ns].v, (short)0, acc[ms][ns], false, false);
#else
    (void)fa; (void)fb;
#endif
    __syncthreads();
    buf ^= 1;
  }

  // ---- epilogue: C layout VGPR i -> M=i (lanes0-15) / M=i+8 (lanes16-31), N=lane%16 ----
  unsigned short* Cb16 = (unsigned short*)Cv + (size_t)blockIdx.z * (size_t)strideCz;
  float*          Cf32 = (float*)Cv          + (size_t)blockIdx.z * (size_t)strideCz;
  for (int ms = 0; ms < 2; ++ms) {
    const int mrow0 = m0 + wm * 32 + ms * 16 + ((lane >> 4) * 8);
    for (int ns = 0; ns < 4; ++ns) {
      const int n  = n0 + wn * 64 + ns * 16 + (lane & 15);
      const float bn = bias ? bias[n] : 0.0f;
      for (int i = 0; i < 8; ++i) {
        const int m = mrow0 + i;
        float v = acc[ms][ns][i] * alpha + bn;
        if (denom) v /= denom[m];
        if (relu)  v = fmaxf(v, 0.0f);
        if (cIsBf16) Cb16[(size_t)m * (size_t)ldc + n] = f2bf(v);
        else         Cf32[(size_t)m * (size_t)ldc + n] = v;
      }
    }
  }
}

// ---------------------------------------------------------------------------
// LayerNorm (unbiased var, eps added to std) -> bf16
// ---------------------------------------------------------------------------
__global__ __launch_bounds__(256) void layernorm_kernel(
    const float* __restrict__ x, const float* __restrict__ a,
    const float* __restrict__ b, unsigned short* __restrict__ y)
{
  const size_t row = blockIdx.x;
  const int tid = threadIdx.x;
  __shared__ float red[256];
  const float* xr = x + row * D_;
  const float x0 = xr[tid], x1 = xr[tid + 256], x2 = xr[tid + 512];
  const float mean = block_reduce_sum(x0 + x1 + x2, red) * (1.0f / D_);
  const float d0 = x0 - mean, d1 = x1 - mean, d2 = x2 - mean;
  const float var = block_reduce_sum(d0 * d0 + d1 * d1 + d2 * d2, red) * (1.0f / (D_ - 1));
  const float inv = 1.0f / (sqrtf(var) + 1e-6f);
  y[row * D_ + tid]       = f2bf(a[tid]       * d0 * inv + b[tid]);
  y[row * D_ + tid + 256] = f2bf(a[tid + 256] * d1 * inv + b[tid + 256]);
  y[row * D_ + tid + 512] = f2bf(a[tid + 512] * d2 * inv + b[tid + 512]);
}

// ---------------------------------------------------------------------------
// Per-batch: softmax over heads' score rows, mean over heads, diag=1,
// row-mask multiply, denom = rowsum+1.  scores: [H,S,S] f32 for one batch.
// ---------------------------------------------------------------------------
__global__ __launch_bounds__(256) void softmax_adj_kernel(
    const float* __restrict__ scores, const int* __restrict__ mask,
    unsigned short* __restrict__ adj, float* __restrict__ denomOut)
{
  const int m = blockIdx.x, tid = threadIdx.x;
  __shared__ float red[256];
  __shared__ float accum[S_];
  accum[tid] = 0.0f; accum[tid + 256] = 0.0f;
  const int mk0 = mask[tid], mk1 = mask[tid + 256];
  __syncthreads();
  for (int h = 0; h < H_; ++h) {
    const float* row = scores + ((size_t)h * S_ + m) * S_;
    const float x0 = mk0 ? row[tid]       : -1e9f;
    const float x1 = mk1 ? row[tid + 256] : -1e9f;
    red[tid] = fmaxf(x0, x1); __syncthreads();
    for (int s = 128; s > 0; s >>= 1) { if (tid < s) red[tid] = fmaxf(red[tid], red[tid + s]); __syncthreads(); }
    const float mx = red[0]; __syncthreads();
    const float e0 = __expf(x0 - mx), e1 = __expf(x1 - mx);
    red[tid] = e0 + e1; __syncthreads();
    for (int s = 128; s > 0; s >>= 1) { if (tid < s) red[tid] += red[tid + s]; __syncthreads(); }
    const float inv = (1.0f / H_) / red[0]; __syncthreads();
    accum[tid] += e0 * inv; accum[tid + 256] += e1 * inv;
    __syncthreads();
  }
  if (tid == 0) accum[m] = 1.0f;          // diagonal := 1
  __syncthreads();
  const float rowm = (float)mask[m];      // adj *= mask[:, :, None]
  const float v0 = accum[tid] * rowm, v1 = accum[tid + 256] * rowm;
  red[tid] = v0 + v1; __syncthreads();
  for (int s = 128; s > 0; s >>= 1) { if (tid < s) red[tid] += red[tid + s]; __syncthreads(); }
  if (tid == 0) denomOut[m] = red[0] + 1.0f;
  adj[(size_t)m * S_ + tid]       = f2bf(v0);
  adj[(size_t)m * S_ + 256 + tid] = f2bf(v1);
}

// ---------------------------------------------------------------------------
// outputs = sum_s out[b,s,:]   (optionally weighted by aspect mask)
// ---------------------------------------------------------------------------
__global__ __launch_bounds__(256) void seq_reduce_kernel(
    const unsigned short* __restrict__ o, const float* __restrict__ aspect,
    float* __restrict__ res)
{
  const int j = blockIdx.x * 256 + threadIdx.x;
  const int b = blockIdx.y;
  const unsigned short* base = o + (size_t)b * S_ * MEM_ + j;
  float s = 0.0f;
  if (aspect) {
    const float* am = aspect + (size_t)b * S_;
    for (int st = 0; st < S_; ++st) s += bf2f(base[(size_t)st * MEM_]) * am[st];
  } else {
    for (int st = 0; st < S_; ++st) s += bf2f(base[(size_t)st * MEM_]);
  }
  res[(size_t)b * MEM_ + j] = s;
}

__global__ void f2bf_kernel(const float* __restrict__ in, unsigned short* __restrict__ out, int n) {
  for (int i = blockIdx.x * blockDim.x + threadIdx.x; i < n; i += gridDim.x * blockDim.x)
    out[i] = f2bf(in[i]);
}

// ---------------------------------------------------------------------------
// Final: aspect normalization, projections, logits, output packing.
// d_out layout: logits_p[32,3] | logits_c[32,3] | fp_y[32,768] | label_emb[32,768]
// ---------------------------------------------------------------------------
__global__ __launch_bounds__(256) void final_kernel(
    const float* __restrict__ redFp, const float* __restrict__ redFcRaw,
    const float* __restrict__ amask, const float* __restrict__ pooled,
    const float* __restrict__ fpdW, const float* __restrict__ fpdb,
    const float* __restrict__ fcdW, const float* __restrict__ fcdb,
    float* __restrict__ out)
{
  const int b = blockIdx.x, tid = threadIdx.x;
  __shared__ float red[256];
  __shared__ float fp[MEM_], fc[MEM_], fy[MEM_];

  const float aspw = block_reduce_sum(amask[(size_t)b * S_ + tid] + amask[(size_t)b * S_ + 256 + tid], red);
  for (int j = tid; j < MEM_; j += 256) {
    fp[j] = redFp[(size_t)b * MEM_ + j];
    fc[j] = redFcRaw[(size_t)b * MEM_ + j] / aspw;
  }
  __syncthreads();

  float p1 = 0.0f, p2 = 0.0f;
  for (int j = tid; j < MEM_; j += 256) { p1 += fp[j] * fc[j]; p2 += fc[j] * fc[j]; }
  const float s_fpfc = block_reduce_sum(p1, red);
  const float s_fcfc = block_reduce_sum(p2, red);
  const float bmo  = sqrtf(s_fcfc);
  const float coef = s_fpfc / bmo;
  const float den  = fmaxf(bmo, 1e-12f);
  for (int j = tid; j < MEM_; j += 256) { fy[j] = fp[j] - coef * (fc[j] / den); }  // r = fp - fp_x
  __syncthreads();

  float p3 = 0.0f, p4 = 0.0f;
  for (int j = tid; j < MEM_; j += 256) { p3 += fp[j] * fy[j]; p4 += fy[j] * fy[j]; }
  const float s_fpr = block_reduce_sum(p3, red);
  const float s_rr  = block_reduce_sum(p4, red);
  const float bmo2  = sqrtf(s_rr);
  const float coef2 = s_fpr / bmo2;
  const float den2  = fmaxf(bmo2, 1e-12f);
  for (int j = tid; j < MEM_; j += 256) {
    const float y = coef2 * (fy[j] / den2);
    fy[j] = y;
    out[192 + (size_t)b * MEM_ + j] = y;                 // fp_y
  }
  __syncthreads();

  for (int c = 0; c < 3; ++c) {
    float q = 0.0f;
    for (int j = tid; j < MEM_; j += 256) q += fy[j] * fpdW[(size_t)c * MEM_ + j];
    const float s = block_reduce_sum(q, red);
    if (tid == 0) out[(size_t)b * 3 + c] = s + fpdb[c];  // logits_p
  }
  for (int c = 0; c < 3; ++c) {
    float q = 0.0f;
    for (int j = tid; j < MEM_; j += 256) q += fc[j] * fcdW[(size_t)c * MEM_ + j];
    const float s = block_reduce_sum(q, red);
    if (tid == 0) out[96 + (size_t)b * 3 + c] = s + fcdb[c];  // logits_c
  }
  for (int j = tid; j < MEM_; j += 256)
    out[24768 + (size_t)b * MEM_ + j] = pooled[(size_t)b * D_ + j];  // label_embedding
}

// ---------------------------------------------------------------------------
extern "C" void kernel_launch(void* const* d_in, const int* in_sizes, int n_in,
                              void* d_out, int out_size, void* d_ws, size_t ws_size,
                              hipStream_t stream)
{
  (void)in_sizes; (void)n_in; (void)out_size; (void)ws_size;
  const float* seq    = (const float*)d_in[0];
  const float* pooled = (const float*)d_in[1];
  const int*   smask  = (const int*)d_in[2];
  const float* amask  = (const float*)d_in[3];
  const float* ln_a   = (const float*)d_in[4];
  const float* ln_b   = (const float*)d_in[5];
  const float* fpWq = (const float*)d_in[6];  const float* fpbq = (const float*)d_in[7];
  const float* fpWk = (const float*)d_in[8];  const float* fpbk = (const float*)d_in[9];
  const float* fpW0 = (const float*)d_in[10]; const float* fpb0 = (const float*)d_in[11];
  const float* fpW1 = (const float*)d_in[12]; const float* fpb1 = (const float*)d_in[13];
  const float* fcWq = (const float*)d_in[14]; const float* fcbq = (const float*)d_in[15];
  const float* fcWk = (const float*)d_in[16]; const float* fcbk = (const float*)d_in[17];
  const float* fcW0 = (const float*)d_in[18]; const float* fcb0 = (const float*)d_in[19];
  const float* fcW1 = (const float*)d_in[20]; const float* fcb1 = (const float*)d_in[21];
  const float* fpdW = (const float*)d_in[22]; const float* fpdb = (const float*)d_in[23];
  const float* fcdW = (const float*)d_in[24]; const float* fcdb = (const float*)d_in[25];

  char* ws = (char*)d_ws;
  size_t off = 0;
  auto carve = [&](size_t bytes) -> void* {
    void* p = ws + off;
    off = (off + bytes + 255) & ~(size_t)255;
    return p;
  };
  unsigned short* Xbf  = (unsigned short*)carve((size_t)B_ * S_ * D_ * 2);
  unsigned short* Wbf  = (unsigned short*)carve((size_t)8 * D_ * D_ * 2);
  unsigned short* Qbf  = (unsigned short*)carve((size_t)B_ * S_ * D_ * 2);
  unsigned short* Kbf  = (unsigned short*)carve((size_t)B_ * S_ * D_ * 2);
  unsigned short* Gbuf = (unsigned short*)carve((size_t)B_ * S_ * D_ * 2);
  unsigned short* Obuf = (unsigned short*)carve((size_t)B_ * S_ * MEM_ * 2);
  float* scoresBuf = (float*)carve((size_t)H_ * S_ * S_ * 4);
  unsigned short* adjBuf = (unsigned short*)carve((size_t)B_ * S_ * S_ * 2);
  float* denomBuf = (float*)carve((size_t)B_ * S_ * 4);
  float* redFp    = (float*)carve((size_t)B_ * MEM_ * 4);
  float* redFc    = (float*)carve((size_t)B_ * MEM_ * 4);

  const dim3 blk(256);

  // LayerNorm -> bf16 X
  layernorm_kernel<<<B_ * S_, blk, 0, stream>>>(seq, ln_a, ln_b, Xbf);

  // Weights -> bf16 (order: fpWq fpWk fpW0 fpW1 fcWq fcWk fcW0 fcW1)
  const float* wsrc[8] = {fpWq, fpWk, fpW0, fpW1, fcWq, fcWk, fcW0, fcW1};
  for (int i = 0; i < 8; ++i)
    f2bf_kernel<<<(D_ * D_) / 256, blk, 0, stream>>>(wsrc[i], Wbf + (size_t)i * D_ * D_, D_ * D_);

  const long long SS  = (long long)S_ * S_;
  const long long SD  = (long long)S_ * D_;
  const float inv_sqrt_dk = 0.10206207261596575f;  // 1/sqrt(96)

  auto run_branch = [&](int widx, const float* bq, const float* bk,
                        const float* b0, const float* b1,
                        float* redOut, bool useAspect)
  {
    const unsigned short* WqB = Wbf + (size_t)(widx + 0) * D_ * D_;
    const unsigned short* WkB = Wbf + (size_t)(widx + 1) * D_ * D_;
    const unsigned short* W0B = Wbf + (size_t)(widx + 2) * D_ * D_;
    const unsigned short* W1B = Wbf + (size_t)(widx + 3) * D_ * D_;

    // Q = X @ Wq^T + bq ; K = X @ Wk^T + bk     (M=16384, N=K=768)
    gemm_bf16_wmma<<<dim3(D_ / 128, (B_ * S_) / 128, 1), blk, 0, stream>>>(
        Xbf, D_, 0, WqB, D_, 0, 1, Qbf, D_, 0, 1, bq, nullptr, D_, 1.0f, 0);
    gemm_bf16_wmma<<<dim3(D_ / 128, (B_ * S_) / 128, 1), blk, 0, stream>>>(
        Xbf, D_, 0, WkB, D_, 0, 1, Kbf, D_, 0, 1, bk, nullptr, D_, 1.0f, 0);

    // Per batch: scores[h] = Q_bh @ K_bh^T / sqrt(DK); then softmax/mean/diag/mask
    for (int b = 0; b < B_; ++b) {
      gemm_bf16_wmma<<<dim3(S_ / 128, S_ / 128, H_), blk, 0, stream>>>(
          Qbf + (size_t)b * S_ * D_, D_, DK_,
          Kbf + (size_t)b * S_ * D_, D_, DK_, 1,
          scoresBuf, S_, SS, 0, nullptr, nullptr, DK_, inv_sqrt_dk, 0);
      softmax_adj_kernel<<<S_, blk, 0, stream>>>(
          scoresBuf, smask + (size_t)b * S_,
          adjBuf + (size_t)b * S_ * S_, denomBuf + (size_t)b * S_);
    }

    // g0 = adj @ X   (batched z=32, A bf16 [S,S], B K-major [S,D])
    gemm_bf16_wmma<<<dim3(D_ / 128, S_ / 128, B_), blk, 0, stream>>>(
        adjBuf, S_, SS, Xbf, D_, SD, 0, Gbuf, D_, SD, 1,
        nullptr, nullptr, S_, 1.0f, 0);
    // out0 = relu((g0 @ W0^T + b0) / denom)
    gemm_bf16_wmma<<<dim3(MEM_ / 128, (B_ * S_) / 128, 1), blk, 0, stream>>>(
        Gbuf, D_, 0, W0B, D_, 0, 1, Obuf, MEM_, 0, 1, b0, denomBuf, D_, 1.0f, 1);
    // g1 = adj @ out0
    gemm_bf16_wmma<<<dim3(MEM_ / 128, S_ / 128, B_), blk, 0, stream>>>(
        adjBuf, S_, SS, Obuf, MEM_, SD, 0, Gbuf, MEM_, SD, 1,
        nullptr, nullptr, S_, 1.0f, 0);
    // out1 = relu((g1 @ W1^T + b1) / denom)
    gemm_bf16_wmma<<<dim3(MEM_ / 128, (B_ * S_) / 128, 1), blk, 0, stream>>>(
        Gbuf, MEM_, 0, W1B, MEM_, 0, 1, Obuf, MEM_, 0, 1, b1, denomBuf, MEM_, 1.0f, 1);

    seq_reduce_kernel<<<dim3(MEM_ / 256, B_), blk, 0, stream>>>(
        Obuf, useAspect ? amask : nullptr, redOut);
  };

  run_branch(0, fpbq, fpbk, fpb0, fpb1, redFp, false);   // fp branch: plain sum
  run_branch(4, fcbq, fcbk, fcb0, fcb1, redFc, true);    // fc branch: aspect-weighted

  final_kernel<<<B_, blk, 0, stream>>>(redFp, redFc, amask, pooled,
                                       fpdW, fpdb, fcdW, fcdb, (float*)d_out);
}
